// tinystoriesgru_45681272160887
// MI455X (gfx1250) — matrix operand order repaired and checked by reading, not compiled, not run on hardware
//
#include <hip/hip_runtime.h>
#include <cstddef>

#define BB 4096
#define TT 256
#define EE 96
#define HH 6
#define VV 98
#define HD 16
#define E3 288

typedef __attribute__((ext_vector_type(16))) _Float16 v16h;
typedef __attribute__((ext_vector_type(8)))  _Float16 v8h;
typedef __attribute__((ext_vector_type(8)))  float    v8f;

// ---- workspace layout -------------------------------------------------------
// f32 tables:  tokQ(V*E) tokK(V*E) tokV(V*E) posK(T*E) posV(T*E) qpos(E)
// f16 weights: out_proj(96*96) gru_ih(288*288) gru_hh(288*96) mw(96*96) head(98*96)
#define WS_F32   (3 * VV * EE + 2 * TT * EE + EE)   // 77472 floats (byte off %32==0)
#define OFF_OUTP 0
#define OFF_IH   (EE * EE)                           // 9216
#define OFF_HH   (OFF_IH + E3 * E3)                  // 92160
#define OFF_MW   (OFF_HH + E3 * EE)                  // 119808
#define OFF_HEAD (OFF_MW + EE * EE)                  // 129024
#define NW_H     (OFF_HEAD + VV * EE)                // 138432 halfs

// ---------------- Precompute 1: fold QKV projection through the embedding gather.
__global__ __launch_bounds__(256) void tsg_precompute(
    const float* __restrict__ tok_emb, const float* __restrict__ pos_emb,
    const float* __restrict__ in_proj_w, const float* __restrict__ in_proj_b,
    float* __restrict__ ws)
{
    const int VE = VV * EE, TE = TT * EE;
    const int ntot = 3 * VE + 2 * TE + EE;
    int tid = blockIdx.x * blockDim.x + threadIdx.x;
    if (tid >= ntot) return;

    const float* src; const float* w; float bias = 0.0f;
    if (tid < 3 * VE) {
        int region = tid / VE;            // 0=Q 1=K 2=V
        int rem = tid - region * VE;
        int v = rem / EE, e = rem % EE;
        src = tok_emb + (size_t)v * EE;
        w   = in_proj_w + (size_t)(region * EE + e) * EE;
    } else if (tid < 3 * VE + 2 * TE) {
        int r2 = tid - 3 * VE;
        int region = r2 / TE;             // 0=K 1=V
        int rem = r2 - region * TE;
        int t = rem / EE, e = rem % EE;
        src  = pos_emb + (size_t)t * EE;
        w    = in_proj_w + (size_t)((region + 1) * EE + e) * EE;
        bias = in_proj_b[(region + 1) * EE + e];
    } else {
        int e = tid - 3 * VE - 2 * TE;
        src  = pos_emb + (size_t)(TT - 1) * EE;
        w    = in_proj_w + (size_t)e * EE;
        bias = in_proj_b[e];
    }
    float acc = bias;
    for (int k = 0; k < EE; ++k) acc += src[k] * w[k];
    ws[tid] = acc;
}

// ---------------- Precompute 2: one-time f32 -> f16 weight conversion (kept (N,K)
// row-major so a 32x16 B-fragment is one 32B contiguous load per lane).
__global__ __launch_bounds__(256) void tsg_convert_weights(
    const float* __restrict__ out_proj_w, const float* __restrict__ gru_w_ih,
    const float* __restrict__ gru_w_hh, const float* __restrict__ mw_w,
    const float* __restrict__ head_w, float* __restrict__ ws)
{
    _Float16* wsh = (_Float16*)(ws + WS_F32);
    int tid = blockIdx.x * blockDim.x + threadIdx.x;
    if (tid >= NW_H) return;
    float v;
    if      (tid < OFF_IH)   v = out_proj_w[tid - OFF_OUTP];
    else if (tid < OFF_HH)   v = gru_w_ih[tid - OFF_IH];
    else if (tid < OFF_MW)   v = gru_w_hh[tid - OFF_HH];
    else if (tid < OFF_HEAD) v = mw_w[tid - OFF_MW];
    else                     v = head_w[tid - OFF_HEAD];
    wsh[tid] = (_Float16)v;
}

// ---------------- WMMA fragment loaders (16-bit 16x32 A / 32x16 B, ISA 7.12.2)
__device__ __forceinline__ v16h load_a_h(const _Float16* A, int lda, int kbase, int lane)
{
    int m = lane & 15;
    int khalf = (lane >> 4) << 4;                 // lanes 16-31: K+16..31
    const _Float16* row = A + m * lda + kbase + khalf;
    v8h lo = *(const v8h*)(row);                  // K = khalf+0..7
    v8h hi = *(const v8h*)(row + 8);              // K = khalf+8..15
    return __builtin_shufflevector(lo, hi, 0,1,2,3,4,5,6,7,8,9,10,11,12,13,14,15);
}

__device__ __forceinline__ v16h load_b_h(const _Float16* W, int ldw, int kbase, int ncol, int lane)
{
    int koff = (lane >> 4) << 4;                  // lanes 16-31: K=16..31
    return *(const v16h*)(W + (size_t)ncol * ldw + kbase + koff);
}

__device__ __forceinline__ v8f wmma_tile(const _Float16* A, int lda,
                                         const _Float16* W, int ldw,
                                         int K, int ncol, int lane)
{
    v8f c = {};
    for (int k = 0; k < K; k += 32) {
        v16h a = load_a_h(A, lda, k, lane);
        v16h b = load_b_h(W, ldw, k, ncol, lane);
        c = __builtin_amdgcn_wmma_f32_16x16x32_f16(false, a, false, b,
                                                   (short)0, c, false, false);
    }
    return c;
}

__device__ __forceinline__ float sigmoidf_(float x) { return 1.0f / (1.0f + __expf(-x)); }

// ---------------- Fused step: one block = 128 threads (4 waves) = 16 batches.
__global__ __launch_bounds__(128) void tsg_fused_step(
    const int* __restrict__ cur_idx, const int* __restrict__ hist,
    const float* __restrict__ hidden, const float* __restrict__ memory,
    const float* __restrict__ tok_emb,
    const float* __restrict__ out_proj_b,
    const float* __restrict__ gru_b_ih, const float* __restrict__ gru_b_hh,
    const float* __restrict__ pg_w, const float* __restrict__ pg_b,
    const float* __restrict__ mw_b,
    const float* __restrict__ ln_g, const float* __restrict__ ln_b,
    const float* __restrict__ head_b,
    const float* __restrict__ ws, float* __restrict__ out)
{
    __shared__ __align__(32) _Float16 shQ[16][EE];   // qh -> ctx -> hidden -> newh (f16 A-staging)
    __shared__ __align__(32) _Float16 shA[16][E3];   // comb = [cur|ctx_o|mem]; later [0:96]=LN out
    __shared__ float shG[16][E3];                    // gi (+ gh for r,z) in f32
    __shared__ float shHn[16][EE];                   // h_n (f32)
    __shared__ float shNh[16][EE];                   // new_hidden (f32, for LN stats / p)
    __shared__ float shRed[4][32][HD];
    __shared__ float shP[16], shMu[16], shRs[16];
    __shared__ int   shIdxLast[16], shCur[16];

    const int tid  = threadIdx.x;
    const int lane = tid & 31;
    const int wav  = tid >> 5;
    const int b0   = blockIdx.x * 16;

    const float* tokQ = ws;
    const float* tokK = tokQ + VV * EE;
    const float* tokV = tokK + VV * EE;
    const float* posK = tokV + VV * EE;
    const float* posV = posK + TT * EE;
    const float* qpos = posV + TT * EE;
    const _Float16* wsh = (const _Float16*)(ws + WS_F32);

    float* out_logits = out;
    float* out_hidden = out + (size_t)BB * VV;
    float* out_memory = out_hidden + (size_t)BB * EE;

    if (tid < 16) {
        shIdxLast[tid] = hist[(size_t)(b0 + tid) * TT + (TT - 1)];
        shCur[tid]     = cur_idx[b0 + tid];
    }
    __syncthreads();

    // qh = tokQ[hist[:,-1]] + qpos ; comb regions: cur_emb, memory
    for (int i = tid; i < 16 * EE; i += 128) {
        int m = i / EE, e = i % EE;
        shQ[m][e]       = (_Float16)(tokQ[(size_t)shIdxLast[m] * EE + e] + qpos[e]);
        shA[m][e]       = (_Float16)tok_emb[(size_t)shCur[m] * EE + e];
        shA[m][192 + e] = (_Float16)memory[(size_t)(b0 + m) * EE + e];
    }
    __syncthreads();

    // ---- attention: each wave owns (m,h) pairs; lanes split T=256 into 8 each
    for (int mh = wav; mh < 16 * HH; mh += 4) {
        int m = mh / HH, h = mh % HH;
        int b = b0 + m;
        float qreg[HD];
#pragma unroll
        for (int d = 0; d < HD; ++d) qreg[d] = (float)shQ[m][h * HD + d];

        float sc[8]; int id8[8];
        float smax = -1e30f;
#pragma unroll
        for (int t8 = 0; t8 < 8; ++t8) {
            int t = t8 * 32 + lane;
            int idx = hist[(size_t)b * TT + t];
            id8[t8] = idx;
            const float* kk = tokK + (size_t)idx * EE + h * HD;
            const float* pk = posK + (size_t)t * EE + h * HD;
            float s = 0.0f;
#pragma unroll
            for (int d = 0; d < HD; ++d) s += qreg[d] * (kk[d] + pk[d]);
            s *= 0.25f;                         // 1/sqrt(HD)
            sc[t8] = s;
            smax = fmaxf(smax, s);
        }
#pragma unroll
        for (int off = 16; off > 0; off >>= 1)
            smax = fmaxf(smax, __shfl_xor(smax, off, 32));
        float ssum = 0.0f;
#pragma unroll
        for (int t8 = 0; t8 < 8; ++t8) { sc[t8] = __expf(sc[t8] - smax); ssum += sc[t8]; }
#pragma unroll
        for (int off = 16; off > 0; off >>= 1)
            ssum += __shfl_xor(ssum, off, 32);
        float inv = 1.0f / ssum;

        float cp[HD];
#pragma unroll
        for (int d = 0; d < HD; ++d) cp[d] = 0.0f;
#pragma unroll
        for (int t8 = 0; t8 < 8; ++t8) {
            int t = t8 * 32 + lane;
            const float* vv = tokV + (size_t)id8[t8] * EE + h * HD;
            const float* pv = posV + (size_t)t * EE + h * HD;
            float wgt = sc[t8] * inv;
#pragma unroll
            for (int d = 0; d < HD; ++d) cp[d] += wgt * (vv[d] + pv[d]);
        }
#pragma unroll
        for (int d = 0; d < HD; ++d) shRed[wav][lane][d] = cp[d];
        __syncthreads();                        // uniform: 24 iters per wave
        if (lane < HD) {
            float acc = 0.0f;
#pragma unroll
            for (int l = 0; l < 32; ++l) acc += shRed[wav][l][lane];
            shQ[m][h * HD + lane] = (_Float16)acc;   // ctx overwrites its own q region
        }
    }
    __syncthreads();

    // ---- ctx @ out_proj^T + b  -> comb[:, 96:192]
    for (int nt = wav; nt < 6; nt += 4) {
        int ncol = nt * 16 + (lane & 15);
        v8f c = wmma_tile(&shQ[0][0], EE, wsh + OFF_OUTP, EE, EE, ncol, lane);
        int rhi = (lane >> 4) << 3;
#pragma unroll
        for (int r = 0; r < 8; ++r)
            shA[rhi + r][96 + ncol] = (_Float16)(c[r] + out_proj_b[ncol]);
    }
    __syncthreads();

    // ---- hidden tile into shQ (f16 copy only feeds the GEMM; gates re-read f32 global)
    for (int i = tid; i < 16 * EE; i += 128) {
        int m = i / EE, e = i % EE;
        shQ[m][e] = (_Float16)hidden[(size_t)(b0 + m) * EE + e];
    }
    __syncthreads();

    // ---- fused per-tile: gi = comb@W_ih^T (K=288) and gh = hidden@W_hh^T (K=96)
    for (int nt = wav; nt < 18; nt += 4) {
        int ncol = nt * 16 + (lane & 15);
        v8f ci = wmma_tile(&shA[0][0], E3, wsh + OFF_IH, E3, E3, ncol, lane);
        v8f ch = wmma_tile(&shQ[0][0], EE, wsh + OFF_HH, EE, EE, ncol, lane);
        int rhi = (lane >> 4) << 3;
        float bi = gru_b_ih[ncol], bh = gru_b_hh[ncol];
        if (ncol < 192) {                    // tile-uniform branch; WMMA already done
#pragma unroll
            for (int r = 0; r < 8; ++r)
                shG[rhi + r][ncol] = ci[r] + bi + ch[r] + bh;
        } else {
#pragma unroll
            for (int r = 0; r < 8; ++r) {
                shG[rhi + r][ncol]        = ci[r] + bi;   // i_n
                shHn[rhi + r][ncol - 192] = ch[r] + bh;   // h_n
            }
        }
    }
    __syncthreads();

    // ---- GRU gates -> new_hidden (f32 math; hidden from global)
    for (int i = tid; i < 16 * EE; i += 128) {
        int m = i / EE, e = i % EE;
        float rg = sigmoidf_(shG[m][e]);
        float zg = sigmoidf_(shG[m][96 + e]);
        float ng = tanhf(shG[m][192 + e] + rg * shHn[m][e]);
        float nh = (1.0f - zg) * ng + zg * hidden[(size_t)(b0 + m) * EE + e];
        shNh[m][e] = nh;
        shQ[m][e]  = (_Float16)nh;           // A-staging for the mw GEMM
        out_hidden[(size_t)(b0 + m) * EE + e] = nh;
    }
    __syncthreads();

    // ---- per-batch scalars: write-gate p, LN mean / rstd (f32)
    if (tid < 16) {
        int m = tid;
        float dot = 0.0f, mu = 0.0f;
        for (int e = 0; e < EE; ++e) { float v = shNh[m][e]; dot += v * pg_w[e]; mu += v; }
        mu *= (1.0f / EE);
        float var = 0.0f;
        for (int e = 0; e < EE; ++e) { float d = shNh[m][e] - mu; var += d * d; }
        var *= (1.0f / EE);
        shP[m]  = sigmoidf_(dot + pg_b[0]);
        shMu[m] = mu;
        shRs[m] = rsqrtf(var + 1e-5f);
    }
    __syncthreads();

    // ---- cand = tanh(newh @ mw^T + b); new_memory = (1-p)*mem + p*cand (mem from global)
    for (int nt = wav; nt < 6; nt += 4) {
        int ncol = nt * 16 + (lane & 15);
        v8f c = wmma_tile(&shQ[0][0], EE, wsh + OFF_MW, EE, EE, ncol, lane);
        int rhi = (lane >> 4) << 3;
        float bb = mw_b[ncol];
#pragma unroll
        for (int r = 0; r < 8; ++r) {
            int m = rhi + r;
            float cand = tanhf(c[r] + bb);
            float p = shP[m];
            out_memory[(size_t)(b0 + m) * EE + ncol] =
                (1.0f - p) * memory[(size_t)(b0 + m) * EE + ncol] + p * cand;
        }
    }

    // ---- LayerNorm into shA[:,0:96] (disjoint from any concurrent reads)
    for (int i = tid; i < 16 * EE; i += 128) {
        int m = i / EE, e = i % EE;
        shA[m][e] = (_Float16)((shNh[m][e] - shMu[m]) * shRs[m] * ln_g[e] + ln_b[e]);
    }
    __syncthreads();

    // ---- logits = LNout @ head^T + b  (N=98 -> 7 tiles, clamp last)
    for (int nt = wav; nt < 7; nt += 4) {
        int ncol = nt * 16 + (lane & 15);
        int nclamp = ncol < VV ? ncol : VV - 1;
        v8f c = wmma_tile(&shA[0][0], E3, wsh + OFF_HEAD, EE, EE, nclamp, lane);
        int rhi = (lane >> 4) << 3;
        if (ncol < VV) {
            float bb = head_b[ncol];
#pragma unroll
            for (int r = 0; r < 8; ++r)
                out_logits[(size_t)(b0 + rhi + r) * VV + ncol] = c[r] + bb;
        }
    }
}

extern "C" void kernel_launch(void* const* d_in, const int* in_sizes, int n_in,
                              void* d_out, int out_size, void* d_ws, size_t ws_size,
                              hipStream_t stream)
{
    (void)in_sizes; (void)n_in; (void)out_size; (void)ws_size;
    const int*   cur_idx    = (const int*)d_in[0];
    const int*   hist       = (const int*)d_in[1];
    const float* hidden     = (const float*)d_in[2];
    const float* memory     = (const float*)d_in[3];
    const float* tok_emb    = (const float*)d_in[4];
    const float* pos_emb    = (const float*)d_in[5];
    const float* in_proj_w  = (const float*)d_in[6];
    const float* in_proj_b  = (const float*)d_in[7];
    const float* out_proj_w = (const float*)d_in[8];
    const float* out_proj_b = (const float*)d_in[9];
    const float* gru_w_ih   = (const float*)d_in[10];
    const float* gru_w_hh   = (const float*)d_in[11];
    const float* gru_b_ih   = (const float*)d_in[12];
    const float* gru_b_hh   = (const float*)d_in[13];
    const float* pg_w       = (const float*)d_in[14];
    const float* pg_b       = (const float*)d_in[15];
    const float* mw_w       = (const float*)d_in[16];
    const float* mw_b       = (const float*)d_in[17];
    const float* ln_g       = (const float*)d_in[18];
    const float* ln_b       = (const float*)d_in[19];
    const float* head_w     = (const float*)d_in[20];
    const float* head_b     = (const float*)d_in[21];
    float* ws   = (float*)d_ws;
    float* outp = (float*)d_out;

    const int ntot = 3 * VV * EE + 2 * TT * EE + EE;
    tsg_precompute<<<(ntot + 255) / 256, 256, 0, stream>>>(
        tok_emb, pos_emb, in_proj_w, in_proj_b, ws);
    tsg_convert_weights<<<(NW_H + 255) / 256, 256, 0, stream>>>(
        out_proj_w, gru_w_ih, gru_w_hh, mw_w, head_w, ws);

    tsg_fused_step<<<BB / 16, 128, 0, stream>>>(
        cur_idx, hist, hidden, memory, tok_emb,
        out_proj_b, gru_b_ih, gru_b_hh,
        pg_w, pg_b, mw_b, ln_g, ln_b, head_b, ws, outp);
}
